// MCPPacmanModel_3659312136241
// MI455X (gfx1250) — matrix operand order repaired and checked by reading, not compiled, not run on hardware
//
#include <hip/hip_runtime.h>
#include <hip/hip_bf16.h>
#include <math.h>

typedef __attribute__((ext_vector_type(16))) __bf16 v16bf;
typedef __attribute__((ext_vector_type(8)))  float  v8f;

#define B_ROWS 16384
#define IN_DIM 2048
#define H_DIM  1024
#define P_N    8
#define A_N    9
#define G_DIM  256
#define NH     (G_DIM + P_N * G_DIM + G_DIM)   // 2560

// ---------------------------------------------------------------- convert f32 -> bf16 (vector x4)
__global__ void k_f32_to_bf16(__bf16* __restrict__ dst, const float* __restrict__ src, long n) {
    long i = ((long)blockIdx.x * blockDim.x + threadIdx.x) * 4;
    if (i + 3 < n) {
        float4 v = *(const float4*)(src + i);
        dst[i + 0] = (__bf16)v.x;
        dst[i + 1] = (__bf16)v.y;
        dst[i + 2] = (__bf16)v.z;
        dst[i + 3] = (__bf16)v.w;
    }
}

// transpose + convert: dst[n*K + k] = src[k*N + n]   (src is [K,N] f32, dst is [N,K] bf16)
__global__ void k_transpose_bf16(__bf16* __restrict__ dst, const float* __restrict__ src,
                                 int K, int N) {
    long idx = (long)blockIdx.x * blockDim.x + threadIdx.x;
    long total = (long)K * N;
    if (idx < total) {
        int n = (int)(idx / K);
        int k = (int)(idx % K);
        dst[idx] = (__bf16)src[(long)k * N + n];
    }
}

// concat head biases: [g_b1(256) | p_b1(2048, p-major) | v_b1(256)]
__global__ void k_bias_concat(float* __restrict__ dst, const float* __restrict__ gb,
                              const float* __restrict__ pb, const float* __restrict__ vb) {
    int i = blockIdx.x * blockDim.x + threadIdx.x;
    if (i < NH) {
        float v;
        if (i < G_DIM)                    v = gb[i];
        else if (i < G_DIM + P_N * G_DIM) v = pb[i - G_DIM];
        else                              v = vb[i - (G_DIM + P_N * G_DIM)];
        dst[i] = v;
    }
}

// ---------------------------------------------------------------- CDNA5 async staging helpers
__device__ __forceinline__ void async_ld16(unsigned lds_off, const void* gaddr) {
    // GLOBAL_LOAD_ASYNC_TO_LDS_B128: 16B memory -> LDS, tracked by ASYNCcnt
    asm volatile("global_load_async_to_lds_b128 %0, %1, off"
                 :: "v"(lds_off), "v"(gaddr) : "memory");
}
__device__ __forceinline__ void wait_async0() {
    asm volatile("s_wait_asynccnt 0" ::: "memory");
}

// ---------------------------------------------------------------- WMMA bf16 GEMM, fused bias+ReLU
// C[M,N] = relu(A[M,K] * Bt[N,K]^T + bias[N]),  A/Bt row-major bf16, C bf16.
#define BM 128
#define BN 256
#define BK 32
#define LDP (BK + 8)   // padded LDS row (halves); 80B rows keep 16B alignment

__global__ __launch_bounds__(256)
void k_gemm_bf16_relu(const __bf16* __restrict__ A, const __bf16* __restrict__ Bt,
                      const float* __restrict__ bias, __bf16* __restrict__ C,
                      int M, int N, int K) {
    __shared__ __bf16 As[2][BM][LDP];   // 20 KB
    __shared__ __bf16 Bs[2][BN][LDP];   // 40 KB

    const int tid  = threadIdx.x;
    const int wave = tid >> 5;
    const int lane = tid & 31;
    const int wm   = wave >> 1;          // 0..3 : 32-row slab
    const int wn   = wave & 1;           // 0..1 : 128-col slab
    const int bm   = blockIdx.y * BM;
    const int bn   = blockIdx.x * BN;

    v8f acc[2][8];
#pragma unroll
    for (int t = 0; t < 2; ++t)
#pragma unroll
        for (int u = 0; u < 8; ++u) acc[t][u] = (v8f)0.0f;

    // Async direct-to-LDS staging: 4096 halves of A + 8192 halves of B per tile.
    auto issueTile = [&](int buf, int k0) {
#pragma unroll
        for (int it = 0; it < 2; ++it) {
            int h = tid * 8 + it * 2048;
            int r = h >> 5, c = h & 31;
            async_ld16((unsigned)(size_t)&As[buf][r][c],
                       (const void*)(A + (size_t)(bm + r) * K + k0 + c));
        }
#pragma unroll
        for (int it = 0; it < 4; ++it) {
            int h = tid * 8 + it * 2048;
            int r = h >> 5, c = h & 31;
            async_ld16((unsigned)(size_t)&Bs[buf][r][c],
                       (const void*)(Bt + (size_t)(bn + r) * K + k0 + c));
        }
    };

    issueTile(0, 0);
    const int KT  = K / BK;
    const int rA  = lane & 15;
    const int kbA = (lane >> 4) * 16;    // A frag: 16 consecutive K per lane
    const int kbB = (lane >> 4) * 16;    // B frag: 16 consecutive K per lane

    for (int kt = 0; kt < KT; ++kt) {
        const int buf = kt & 1;
        wait_async0();                   // my async writes to `buf` complete
        __syncthreads();                 // all waves' writes to `buf` visible; `buf^1` reads done
        if (kt + 1 < KT) issueTile(buf ^ 1, (kt + 1) * BK);

        v16bf afr[2], bfr[8];
#pragma unroll
        for (int t = 0; t < 2; ++t) {
            int row = wm * 32 + t * 16 + rA;
            *((uint4*)&afr[t] + 0) = *(const uint4*)&As[buf][row][kbA];
            *((uint4*)&afr[t] + 1) = *(const uint4*)&As[buf][row][kbA + 8];
        }
#pragma unroll
        for (int u = 0; u < 8; ++u) {
            int col = wn * 128 + u * 16 + rA;
            *((uint4*)&bfr[u] + 0) = *(const uint4*)&Bs[buf][col][kbB];
            *((uint4*)&bfr[u] + 1) = *(const uint4*)&Bs[buf][col][kbB + 8];
        }
#pragma unroll
        for (int t = 0; t < 2; ++t)
#pragma unroll
            for (int u = 0; u < 8; ++u)
                acc[t][u] = __builtin_amdgcn_wmma_f32_16x16x32_bf16(
                    false, afr[t], false, bfr[u], (short)0, acc[t][u], false, false);
    }

    // epilogue: C/D layout -> row = r + 8*(lane>>4), col = lane&15
    const int rofs = 8 * (lane >> 4);
    const int cofs = lane & 15;
#pragma unroll
    for (int t = 0; t < 2; ++t)
#pragma unroll
        for (int u = 0; u < 8; ++u) {
            int colb = bn + wn * 128 + u * 16 + cofs;
            float bsv = bias[colb];
#pragma unroll
            for (int r = 0; r < 8; ++r) {
                int row = bm + wm * 32 + t * 16 + rofs + r;
                float v = acc[t][u][r] + bsv;
                v = fmaxf(v, 0.0f);
                C[(size_t)row * N + colb] = (__bf16)v;
            }
        }
}

// ---------------------------------------------------------------- tail: gate/prim/value fuse
__device__ inline float wred32(float x) {
#pragma unroll
    for (int m = 16; m > 0; m >>= 1) x += __shfl_xor(x, m, 32);
    return x;
}

__global__ __launch_bounds__(256)
void k_heads_tail(const __bf16* __restrict__ HO,          // [B, NH]
                  const float* __restrict__ g_w2,         // [256,8]
                  const float* __restrict__ g_b2,         // [8]
                  const float* __restrict__ p_w2,         // [8,256,9]
                  const float* __restrict__ p_b2,         // [8,9]
                  const float* __restrict__ v_w2,         // [256]
                  const float* __restrict__ v_b2,         // [1]
                  float* __restrict__ actor, float* __restrict__ value, int rows) {
    __shared__ float gw[G_DIM * P_N];
    __shared__ float vw[G_DIM];
    for (int i = threadIdx.x; i < G_DIM * P_N; i += 256) gw[i] = g_w2[i];
    if (threadIdx.x < G_DIM) vw[threadIdx.x] = v_w2[threadIdx.x];
    __syncthreads();

    const int wave = threadIdx.x >> 5;
    const int lane = threadIdx.x & 31;
    const int row  = blockIdx.x * 8 + wave;
    if (row >= rows) return;
    const __bf16* hrow = HO + (size_t)row * NH;

    float gv[8], vv[8];
#pragma unroll
    for (int i = 0; i < 8; ++i) {
        int g = lane + 32 * i;
        gv[i] = (float)hrow[g];
        vv[i] = (float)hrow[G_DIM + P_N * G_DIM + g];
    }

    float gate[8];
#pragma unroll
    for (int p = 0; p < 8; ++p) {
        float s = 0.0f;
#pragma unroll
        for (int i = 0; i < 8; ++i) s += gv[i] * gw[(lane + 32 * i) * P_N + p];
        s = wred32(s) + g_b2[p];
        gate[p] = 1.0f / (1.0f + expf(-s));
    }

    float sv = 0.0f;
#pragma unroll
    for (int i = 0; i < 8; ++i) sv += vv[i] * vw[lane + 32 * i];
    sv = wred32(sv) + v_b2[0];

    float acc9[9];
#pragma unroll
    for (int a = 0; a < 9; ++a) acc9[a] = 0.0f;

#pragma unroll
    for (int p = 0; p < 8; ++p) {
        float ph[8];
#pragma unroll
        for (int i = 0; i < 8; ++i)
            ph[i] = (float)hrow[G_DIM + p * G_DIM + lane + 32 * i];
#pragma unroll
        for (int a = 0; a < 9; ++a) {
            float s = 0.0f;
#pragma unroll
            for (int i = 0; i < 8; ++i)
                s += ph[i] * p_w2[((size_t)(p * G_DIM + lane + 32 * i)) * A_N + a];
            s = wred32(s) + p_b2[p * A_N + a];
            acc9[a] += gate[p] * s;
        }
    }

    if (lane == 0) {
#pragma unroll
        for (int a = 0; a < 9; ++a) actor[(size_t)row * A_N + a] = acc9[a];
        value[row] = sv;
    }
}

// ---------------------------------------------------------------- host
extern "C" void kernel_launch(void* const* d_in, const int* in_sizes, int n_in,
                              void* d_out, int out_size, void* d_ws, size_t ws_size,
                              hipStream_t stream) {
    const float* features = (const float*)d_in[0];
    const float* fe_w1 = (const float*)d_in[1];
    const float* fe_b1 = (const float*)d_in[2];
    const float* fe_w2 = (const float*)d_in[3];
    const float* fe_b2 = (const float*)d_in[4];
    const float* g_w1  = (const float*)d_in[5];
    const float* g_b1  = (const float*)d_in[6];
    const float* g_w2  = (const float*)d_in[7];
    const float* g_b2  = (const float*)d_in[8];
    const float* p_w1  = (const float*)d_in[9];
    const float* p_b1  = (const float*)d_in[10];
    const float* p_w2  = (const float*)d_in[11];
    const float* p_b2  = (const float*)d_in[12];
    const float* v_w1  = (const float*)d_in[13];
    const float* v_b1  = (const float*)d_in[14];
    const float* v_w2  = (const float*)d_in[15];
    const float* v_b2  = (const float*)d_in[16];

    char* ws = (char*)d_ws;
    size_t o = 0;
    __bf16* W1t   = (__bf16*)(ws + o); o += (size_t)H_DIM * IN_DIM * 2;        // [1024,2048]
    __bf16* W2t   = (__bf16*)(ws + o); o += (size_t)H_DIM * H_DIM * 2;         // [1024,1024]
    __bf16* HEADS = (__bf16*)(ws + o); o += (size_t)NH * H_DIM * 2;            // [2560,1024]
    float*  hbias = (float*)(ws + o);  o += (size_t)NH * 4;
    o = (o + 255) & ~(size_t)255;
    size_t o_act = o;                                                          // activation region
    __bf16* Xbf  = (__bf16*)(ws + o); o += (size_t)B_ROWS * IN_DIM * 2;        // 64 MB
    __bf16* H1   = (__bf16*)(ws + o); o += (size_t)B_ROWS * H_DIM * 2;         // 32 MB
    __bf16* Hbf  = (__bf16*)(ws + o); o += (size_t)B_ROWS * H_DIM * 2;         // 32 MB
    __bf16* HOUT = (__bf16*)(ws + o_act);   // alias over Xbf+H1 (both dead by GEMM3)

    // 1) convert / transpose weights + activations to bf16
    {
        long n = (long)B_ROWS * IN_DIM;
        k_f32_to_bf16<<<(unsigned)(n / 1024), 256, 0, stream>>>(Xbf, features, n);
    }
    {
        long t = (long)IN_DIM * H_DIM;
        k_transpose_bf16<<<(unsigned)((t + 255) / 256), 256, 0, stream>>>(W1t, fe_w1, IN_DIM, H_DIM);
    }
    {
        long t = (long)H_DIM * H_DIM;
        k_transpose_bf16<<<(unsigned)((t + 255) / 256), 256, 0, stream>>>(W2t, fe_w2, H_DIM, H_DIM);
    }
    {
        long t = (long)H_DIM * G_DIM;
        unsigned g = (unsigned)((t + 255) / 256);
        k_transpose_bf16<<<g, 256, 0, stream>>>(HEADS, g_w1, H_DIM, G_DIM);
        for (int p = 0; p < P_N; ++p)
            k_transpose_bf16<<<g, 256, 0, stream>>>(HEADS + (size_t)(G_DIM + p * G_DIM) * H_DIM,
                                                    p_w1 + (size_t)p * H_DIM * G_DIM, H_DIM, G_DIM);
        k_transpose_bf16<<<g, 256, 0, stream>>>(HEADS + (size_t)(G_DIM + P_N * G_DIM) * H_DIM,
                                                v_w1, H_DIM, G_DIM);
    }
    k_bias_concat<<<(NH + 255) / 256, 256, 0, stream>>>(hbias, g_b1, p_b1, v_b1);

    // 2) three WMMA GEMMs (block 128x256, async direct-to-LDS staging)
    k_gemm_bf16_relu<<<dim3(H_DIM / BN, B_ROWS / BM), 256, 0, stream>>>(
        Xbf, W1t, fe_b1, H1, B_ROWS, H_DIM, IN_DIM);
    k_gemm_bf16_relu<<<dim3(H_DIM / BN, B_ROWS / BM), 256, 0, stream>>>(
        H1, W2t, fe_b2, Hbf, B_ROWS, H_DIM, H_DIM);
    k_gemm_bf16_relu<<<dim3(NH / BN, B_ROWS / BM), 256, 0, stream>>>(
        Hbf, HEADS, hbias, HOUT, B_ROWS, NH, H_DIM);

    // 3) fused tail -> actor [B,9] then value [B,1] in d_out
    float* actor = (float*)d_out;
    float* value = (float*)d_out + (size_t)B_ROWS * A_N;
    k_heads_tail<<<B_ROWS / 8, 256, 0, stream>>>(HOUT, g_w2, g_b2, p_w2, p_b2,
                                                 v_w2, v_b2, actor, value, B_ROWS);
}
